// ArDiffusion_79697413144597
// MI455X (gfx1250) — matrix operand order repaired and checked by reading, not compiled, not run on hardware
//
#include <hip/hip_runtime.h>
#include <hip/hip_bf16.h>

typedef __attribute__((ext_vector_type(16))) __bf16 v16bf;
typedef __attribute__((ext_vector_type(8)))  __bf16 v8bf;
typedef __attribute__((ext_vector_type(8)))  float  v8f;

// Problem constants (match reference)
constexpr int NSTEP = 8;
constexpr int DSTEP = 128;
constexpr int NE    = 1024;          // NSTEP*DSTEP
constexpr int VOC   = 50257;
constexpr int BATCH = 2;
constexpr int TLEN  = 2039;
constexpr int KLEN  = 2048;          // (TLEN + 2*NSTEP) - (NSTEP-1)
constexpr int ROWS  = BATCH * KLEN;  // 4096
constexpr int NROWS_PRED = BATCH * TLEN; // 4078
constexpr int EMB_BLOCKS = 512;

// Async global->LDS staging helpers (CDNA5; ASYNCcnt-tracked DMA, no VGPRs)
__device__ __forceinline__ void async_g2l_b128(const __bf16* g, const __bf16* l) {
    unsigned lds_off = (unsigned)(uintptr_t)l;          // low 32 bits = LDS addr
    unsigned long long ga = (unsigned long long)(uintptr_t)g;
    asm volatile("global_load_async_to_lds_b128 %0, %1, off"
                 :: "v"(lds_off), "v"(ga) : "memory");
}
__device__ __forceinline__ void wait_async0() {
    asm volatile("s_wait_asynccnt 0x0" ::: "memory");
}

// ---------------------------------------------------------------------------
// 1) Build x (tilted embeddings + noise + wpe), emit bf16
// ---------------------------------------------------------------------------
__global__ void build_x_kernel(const float* __restrict__ noise,
                               const float* __restrict__ leftn,
                               const float* __restrict__ rightn,
                               const float* __restrict__ wte,
                               const float* __restrict__ wpe,
                               const int*   __restrict__ toks,
                               __bf16* __restrict__ xb) {
    long idx = (long)blockIdx.x * blockDim.x + threadIdx.x;   // row*NE + c
    if (idx >= (long)ROWS * NE) return;
    int  c   = (int)(idx & (NE - 1));
    long row = idx >> 10;              // b*KLEN + k
    int  b   = (int)(row >> 11);
    int  k   = (int)(row & (KLEN - 1));
    int  r   = c >> 7;                 // step
    int  d   = c & (DSTEP - 1);
    int  s   = k + r;                  // index into cat axis (size TLEN+16)
    float v;
    if (s < NSTEP) {
        v = leftn[((b * NSTEP + s) * NSTEP + r) * DSTEP + d];
    } else if (s < NSTEP + TLEN) {
        int   t   = s - NSTEP;
        float w   = (float)(r + 1) * 0.125f;
        int   tok = toks[b * TLEN + t];
        float e   = wte[(long)tok * DSTEP + d];
        float n   = noise[(((long)b * TLEN + t) * NSTEP + r) * DSTEP + d];
        v = e * (1.0f - w) + n * w;
    } else {
        int j = s - NSTEP - TLEN;
        v = rightn[((b * NSTEP + j) * NSTEP + r) * DSTEP + d];
    }
    v += wpe[(long)k * NE + c];
    xb[idx] = (__bf16)v;
}

// ---------------------------------------------------------------------------
// 2) Convert wte -> bf16 (row-major, [VOC][128])
// ---------------------------------------------------------------------------
__global__ void cvt_wte_kernel(const float* __restrict__ wte,
                               __bf16* __restrict__ wteb) {
    long idx = (long)blockIdx.x * blockDim.x + threadIdx.x;
    if (idx >= (long)VOC * DSTEP) return;
    wteb[idx] = (__bf16)wte[idx];
}

// ---------------------------------------------------------------------------
// 3) Transpose W_bb (d,e) -> W_bb^T (e,d) in bf16 so B-fragments load
//    contiguous K runs.
// ---------------------------------------------------------------------------
__global__ void transp_wbb_kernel(const float* __restrict__ wbb,
                                  __bf16* __restrict__ wbbT) {
    long idx = (long)blockIdx.x * blockDim.x + threadIdx.x;  // e*NE + d
    if (idx >= (long)NE * NE) return;
    int e = (int)(idx >> 10);
    int d = (int)(idx & (NE - 1));
    wbbT[idx] = (__bf16)wbb[(long)d * NE + e];
}

// ---------------------------------------------------------------------------
// 4) GEMM1: new_x = tanh(x @ W_bb).  A:[4096][1024]bf16, B^T:[1024][1024]bf16.
//    8 waves/block; wave -> 16 rows x 64 cols (4 accumulators).
//    B panel shared by all waves: double-buffered K-chunks (128) staged into
//    LDS via async DMA, padded row stride kills bank conflicts.
// ---------------------------------------------------------------------------
constexpr int H_KC   = 128;                 // K elems per staged chunk
constexpr int H_LDST = H_KC + 8;            // padded row stride (elems)
constexpr int H_NCH  = NE / H_KC;           // 8 chunks

__global__ void __launch_bounds__(256)
gemm_hidden_kernel(const __bf16* __restrict__ xb,
                   const __bf16* __restrict__ wbbT,
                   float*  __restrict__ newx,
                   __bf16* __restrict__ topb) {
    __shared__ __attribute__((aligned(32))) __bf16 ldsB[2][64 * H_LDST]; // 34.8 KB
    const int tid  = threadIdx.x;
    const int lane = tid & 31;
    const int wave = tid >> 5;
    const int kh   = lane >> 4;      // K-half selector
    const int ln   = lane & 15;      // A row / B col within tile
    const int rowBase = blockIdx.y * 128 + wave * 16;
    const int colBase = blockIdx.x * 64;

    // Per-thread staging geometry: 64 rows x 128 elems = 1024 x 16B, 4/thread
    const int st_row = (tid + 0 * 256) >> 4;   // recomputed per it below

    v8f acc[4];
    acc[0] = {}; acc[1] = {}; acc[2] = {}; acc[3] = {};

    // Prologue: stage chunk 0 into buffer 0
#pragma unroll
    for (int it = 0; it < 4; ++it) {
        int ch  = tid + it * 256;
        int row = ch >> 4;
        int ko  = (ch & 15) * 8;
        async_g2l_b128(wbbT + (size_t)(colBase + row) * NE + ko,
                       &ldsB[0][row * H_LDST + ko]);
    }
    wait_async0();
    __syncthreads();

    for (int kc = 0; kc < H_NCH; ++kc) {
        const int cur   = kc & 1;
        const int kbase = kc * H_KC;
        // Issue DMA for next chunk into the other buffer (no wait yet)
        if (kc + 1 < H_NCH) {
            const int knext = (kc + 1) * H_KC;
#pragma unroll
            for (int it = 0; it < 4; ++it) {
                int ch  = tid + it * 256;
                int row = ch >> 4;
                int ko  = (ch & 15) * 8;
                async_g2l_b128(wbbT + (size_t)(colBase + row) * NE + knext + ko,
                               &ldsB[cur ^ 1][row * H_LDST + ko]);
            }
        }
        // Compute current chunk: 4 K-steps of 32
#pragma unroll
        for (int ks = 0; ks < H_KC / 32; ++ks) {
            const int k0 = ks * 32;
            union { v16bf v; v8bf h[2]; } a;
            const __bf16* ap = xb + (size_t)(rowBase + ln) * NE + kbase + k0 + kh * 8;
            a.h[0] = *(const v8bf*)(ap);
            a.h[1] = *(const v8bf*)(ap + 16);
            // Load ALL four B fragments first: one dscnt wait, 4 WMMAs b2b
            union { v16bf v; v8bf h[2]; } b[4];
#pragma unroll
            for (int nt = 0; nt < 4; ++nt) {
                const __bf16* lb = &ldsB[cur][(ln + 16 * nt) * H_LDST + k0 + kh * 8];
                b[nt].h[0] = *(const v8bf*)(lb);
                b[nt].h[1] = *(const v8bf*)(lb + 16);
            }
#pragma unroll
            for (int nt = 0; nt < 4; ++nt) {
                acc[nt] = __builtin_amdgcn_wmma_f32_16x16x32_bf16(
                    false, a.v, false, b[nt].v, (short)0, acc[nt], false, false);
            }
        }
        // My next-chunk DMA done; barrier also guarantees all waves finished
        // reading `cur` before anyone overwrites it in the next iteration.
        wait_async0();
        __syncthreads();
    }

#pragma unroll
    for (int nt = 0; nt < 4; ++nt) {
        const int col = colBase + nt * 16 + ln;
#pragma unroll
        for (int i = 0; i < 8; ++i) {
            const int row = rowBase + kh * 8 + i;
            float v = tanhf(acc[nt][i]);
            newx[(size_t)row * NE + col] = v;
            if (col < DSTEP) topb[(size_t)row * DSTEP + col] = (__bf16)v;
        }
    }
}

// ---------------------------------------------------------------------------
// 5) GEMM2: tok_logits = topmost @ wte^T.  A:[4096][128]bf16, B rows = wte_b.
//    Memory-bound on the 823MB fp32 output. 64-vocab-col x K=128 wte panel
//    (17KB) async-staged to LDS once per block, shared by all 8 waves.
// ---------------------------------------------------------------------------
constexpr int L_LDST = DSTEP + 8;           // padded row stride (elems)

__global__ void __launch_bounds__(256)
gemm_logits_kernel(const __bf16* __restrict__ topb,
                   const __bf16* __restrict__ wteb,
                   float* __restrict__ out) {
    __shared__ __attribute__((aligned(32))) __bf16 ldsW[64 * L_LDST]; // 17 KB
    const int tid  = threadIdx.x;
    const int lane = tid & 31;
    const int wave = tid >> 5;
    const int kh   = lane >> 4;
    const int ln   = lane & 15;
    const int rowBase = blockIdx.y * 128 + wave * 16;
    const int colBase = blockIdx.x * 64;

    // Stage 64 wte rows (clamped at vocab edge): 1024 x 16B chunks, 4/thread.
#pragma unroll
    for (int it = 0; it < 4; ++it) {
        int ch   = tid + it * 256;
        int row  = ch >> 4;                 // 0..63
        int ko   = (ch & 15) * 8;           // elem offset in row
        int grow = colBase + row;
        grow = (grow < VOC) ? grow : (VOC - 1);   // branchless clamp
        async_g2l_b128(wteb + (size_t)grow * DSTEP + ko,
                       &ldsW[row * L_LDST + ko]);
    }
    wait_async0();
    __syncthreads();

    v8f acc[4];
    acc[0] = {}; acc[1] = {}; acc[2] = {}; acc[3] = {};

#pragma unroll
    for (int ks = 0; ks < 4; ++ks) {
        const int k0 = ks * 32;
        union { v16bf v; v8bf h[2]; } a;
        const __bf16* ap = topb + (size_t)(rowBase + ln) * DSTEP + k0 + kh * 8;
        a.h[0] = *(const v8bf*)(ap);
        a.h[1] = *(const v8bf*)(ap + 16);
        union { v16bf v; v8bf h[2]; } b[4];
#pragma unroll
        for (int nt = 0; nt < 4; ++nt) {
            const __bf16* lb = &ldsW[(ln + 16 * nt) * L_LDST + k0 + kh * 8];
            b[nt].h[0] = *(const v8bf*)(lb);
            b[nt].h[1] = *(const v8bf*)(lb + 16);
        }
#pragma unroll
        for (int nt = 0; nt < 4; ++nt) {
            acc[nt] = __builtin_amdgcn_wmma_f32_16x16x32_bf16(
                false, a.v, false, b[nt].v, (short)0, acc[nt], false, false);
        }
    }

#pragma unroll
    for (int nt = 0; nt < 4; ++nt) {
        const int col = colBase + nt * 16 + ln;
        if (col < VOC) {
#pragma unroll
            for (int i = 0; i < 8; ++i) {
                const int row = rowBase + kh * 8 + i;
                out[(size_t)row * VOC + col] = acc[nt][i];
            }
        }
    }
}

// ---------------------------------------------------------------------------
// 6) emb_loss partials: sum (new_x[:, :-1] - new_x[:, 1:])^2, fixed-order.
// ---------------------------------------------------------------------------
__global__ void emb_partial_kernel(const float* __restrict__ newx,
                                   float* __restrict__ partial) {
    const long total = (long)BATCH * (KLEN - 1) * NE;   // 2*2047*1024
    float acc = 0.0f;
    for (long i = (long)blockIdx.x * blockDim.x + threadIdx.x; i < total;
         i += (long)EMB_BLOCKS * 256) {
        long b   = i / ((long)(KLEN - 1) * NE);
        long rkd = i % ((long)(KLEN - 1) * NE);
        long k   = rkd >> 10;
        long e   = rkd & (NE - 1);
        long o   = (b * KLEN + k) * NE + e;
        float d  = newx[o] - newx[o + NE];
        acc += d * d;
    }
    __shared__ float sm[256];
    sm[threadIdx.x] = acc;
    __syncthreads();
    for (int off = 128; off > 0; off >>= 1) {
        if ((int)threadIdx.x < off) sm[threadIdx.x] += sm[threadIdx.x + off];
        __syncthreads();
    }
    if (threadIdx.x == 0) partial[blockIdx.x] = sm[0];
}

// ---------------------------------------------------------------------------
// 7) Per-row online logsumexp + NLL (one block per pred row).
// ---------------------------------------------------------------------------
__global__ void nll_kernel(const float* __restrict__ out,
                           const int*   __restrict__ toks,
                           float* __restrict__ nllArr) {
    const int j  = blockIdx.x;          // 0..4077
    const int b  = j / TLEN;
    const int tt = j - b * TLEN;
    const float* row = out + (size_t)(b * KLEN + tt + NSTEP) * VOC;

    float m = -3.402823466e38f, s = 0.0f;
    for (int v = threadIdx.x; v < VOC; v += 256) {
        float x = row[v];
        if (x > m) { s = s * __expf(m - x) + 1.0f; m = x; }
        else       { s += __expf(x - m); }
    }
    __shared__ float sm[256], ss[256];
    sm[threadIdx.x] = m;
    ss[threadIdx.x] = s;
    __syncthreads();
    for (int off = 128; off > 0; off >>= 1) {
        if ((int)threadIdx.x < off) {
            float m1 = sm[threadIdx.x],       s1 = ss[threadIdx.x];
            float m2 = sm[threadIdx.x + off], s2 = ss[threadIdx.x + off];
            float mm = fmaxf(m1, m2);
            sm[threadIdx.x] = mm;
            ss[threadIdx.x] = s1 * __expf(m1 - mm) + s2 * __expf(m2 - mm);
        }
        __syncthreads();
    }
    if (threadIdx.x == 0) {
        float lse = sm[0] + __logf(ss[0]);
        int   tgt = toks[b * TLEN + tt];
        nllArr[j] = lse - row[tgt];
    }
}

// ---------------------------------------------------------------------------
// 8) Final: loss = mean(nll) + mean(diff^2) -> d_out[last]
// ---------------------------------------------------------------------------
__global__ void final_loss_kernel(const float* __restrict__ nllArr,
                                  const float* __restrict__ embPartial,
                                  float* __restrict__ out, long lossIdx) {
    float a = 0.0f, e = 0.0f;
    for (int i = threadIdx.x; i < NROWS_PRED; i += 256) a += nllArr[i];
    for (int i = threadIdx.x; i < EMB_BLOCKS;  i += 256) e += embPartial[i];
    __shared__ float sa[256], se[256];
    sa[threadIdx.x] = a;
    se[threadIdx.x] = e;
    __syncthreads();
    for (int off = 128; off > 0; off >>= 1) {
        if ((int)threadIdx.x < off) {
            sa[threadIdx.x] += sa[threadIdx.x + off];
            se[threadIdx.x] += se[threadIdx.x + off];
        }
        __syncthreads();
    }
    if (threadIdx.x == 0) {
        float logit_loss = sa[0] / (float)NROWS_PRED;
        float emb_loss   = se[0] / ((float)BATCH * (float)(KLEN - 1) * (float)NE);
        out[lossIdx] = logit_loss + emb_loss;
    }
}

// ---------------------------------------------------------------------------
// Launch
// ---------------------------------------------------------------------------
extern "C" void kernel_launch(void* const* d_in, const int* in_sizes, int n_in,
                              void* d_out, int out_size, void* d_ws, size_t ws_size,
                              hipStream_t stream) {
    const float* noise  = (const float*)d_in[0];
    const float* leftn  = (const float*)d_in[1];
    const float* rightn = (const float*)d_in[2];
    const float* wte    = (const float*)d_in[3];
    const float* wpe    = (const float*)d_in[4];
    const float* wbb    = (const float*)d_in[5];
    const int*   toks   = (const int*)d_in[6];
    float* out = (float*)d_out;

    // Workspace layout (256B aligned slabs), ~41.2 MB total
    char* ws = (char*)d_ws;
    __bf16* xb   = (__bf16*)(ws + 0);          //  8,388,608  x bf16 [4096][1024]
    __bf16* wbbT = (__bf16*)(ws + 8388608);    //  2,097,152  W_bb^T bf16 [1024][1024]
    __bf16* wteb = (__bf16*)(ws + 10485760);   // 12,865,792  wte bf16 [VOC][128]
    float*  newx = (float*)(ws + 23351552);    // 16,777,216  new_x f32 [4096][1024]
    __bf16* topb = (__bf16*)(ws + 40128768);   //  1,048,576  topmost bf16 [4096][128]
    float*  nll  = (float*)(ws + 41177344);    //     16,312  per-row nll
    float*  embp = (float*)(ws + 41193728);    //      2,048  emb partials

    build_x_kernel<<<(ROWS * NE + 255) / 256, 256, 0, stream>>>(
        noise, leftn, rightn, wte, wpe, toks, xb);
    cvt_wte_kernel<<<((long)VOC * DSTEP + 255) / 256, 256, 0, stream>>>(wte, wteb);
    transp_wbb_kernel<<<(NE * NE + 255) / 256, 256, 0, stream>>>(wbb, wbbT);

    gemm_hidden_kernel<<<dim3(NE / 64, ROWS / 128), 256, 0, stream>>>(
        xb, wbbT, newx, topb);
    gemm_logits_kernel<<<dim3((VOC + 63) / 64, ROWS / 128), 256, 0, stream>>>(
        topb, wteb, out);

    emb_partial_kernel<<<EMB_BLOCKS, 256, 0, stream>>>(newx, embp);
    nll_kernel<<<NROWS_PRED, 256, 0, stream>>>(out, toks, nll);
    final_loss_kernel<<<1, 256, 0, stream>>>(nll, embp, out, (long)out_size - 1);
}
